// MultiHeadAttention_4801773437352
// MI455X (gfx1250) — compile-verified, matrix operand order checked by reading, not run on hardware
//
#include <hip/hip_runtime.h>
#include <hip/hip_bf16.h>
#include <stdint.h>

#define DMODEL 1024
#define NHEADS 16
#define DK     64
#define BB     4
#define LL     2048

typedef __attribute__((ext_vector_type(16))) __bf16 v16bf;
typedef __attribute__((ext_vector_type(8)))  float  v8f;
typedef int          v4i  __attribute__((ext_vector_type(4)));
typedef unsigned int u32x4 __attribute__((ext_vector_type(4)));
typedef int          i32x4 __attribute__((ext_vector_type(4)));
typedef int          i32x8 __attribute__((ext_vector_type(8)));

union Frag16 { uint4 u[2]; v16bf v; };

__device__ __forceinline__ v16bf load_bf16_frag(const __bf16* c0, const __bf16* c1) {
  Frag16 f;
  f.u[0] = *(const uint4*)c0;   // K-chunk (l/16)*8 .. +7    -> VGPRs 0-3
  f.u[1] = *(const uint4*)c1;   // K-chunk (l/16)*8+16 ..+23 -> VGPRs 4-7
  return f.v;
}

__device__ __forceinline__ v16bf load_f32_frag(const float* c0, const float* c1) {
  v16bf r;
#pragma unroll
  for (int i = 0; i < 8; ++i) r[i] = (__bf16)c0[i];
#pragma unroll
  for (int i = 0; i < 8; ++i) r[8 + i] = (__bf16)c1[i];
  return r;
}

__device__ __forceinline__ v8f wmma_bf16(v16bf a, v16bf b, v8f c) {
  return __builtin_amdgcn_wmma_f32_16x16x32_bf16(false, a, false, b, (short)0, c,
                                                 false, false);
}

// ---------------------------------------------------------------------------
// CDNA5 async data-mover paths (guarded: toolchains differ per bridge doc)
// ---------------------------------------------------------------------------
#define HAS_ASYNC_LDS __has_builtin(__builtin_amdgcn_global_load_async_to_lds_b128)
#define HAS_TDM       __has_builtin(__builtin_amdgcn_tensor_load_to_lds)

__device__ __forceinline__ void async_copy16(__bf16* lds, const __bf16* g) {
#if HAS_ASYNC_LDS
  __builtin_amdgcn_global_load_async_to_lds_b128(
      (__attribute__((address_space(1))) v4i*)(void*)g,
      (__attribute__((address_space(3))) v4i*)(void*)lds, 0, 0);
#else
  *(uint4*)lds = *(const uint4*)g;
#endif
}

__device__ __forceinline__ void wait_async_lds() {
#if HAS_ASYNC_LDS
#if __has_builtin(__builtin_amdgcn_s_wait_asynccnt)
  __builtin_amdgcn_s_wait_asynccnt(0);
#else
  asm volatile("s_wait_asynccnt 0" ::: "memory");
#endif
#endif
}

#if HAS_TDM
// 2-D bf16 tile load via Tensor Data Mover: tile_x contiguous elems per row,
// tile_y rows with row stride `stride` elems; packed into LDS row-major.
__device__ __forceinline__ void tdm_load_tile_bf16(__bf16* lds_dst,
                                                   const __bf16* gsrc,
                                                   unsigned tile_x, unsigned tile_y,
                                                   unsigned stride, unsigned dim_x,
                                                   unsigned dim_y) {
  unsigned long long ga = (unsigned long long)(uintptr_t)gsrc;
  unsigned lds_off =
      (unsigned)(uintptr_t)(__attribute__((address_space(3))) void*)(void*)lds_dst;
  u32x4 g0;
  g0[0] = 1u;                                 // count=1, user-mode descriptor
  g0[1] = lds_off;                            // lds_addr (bytes)
  g0[2] = (unsigned)ga;                       // global_addr[31:0]
  g0[3] = (unsigned)(ga >> 32) | (2u << 30);  // global_addr[56:32] | type=2
  i32x8 g1;
  g1[0] = 1 << 16;                                // data_size=1 -> 2B elems
  g1[1] = (int)(dim_x << 16);                     // tensor_dim0[15:0]
  g1[2] = (int)((dim_x >> 16) | (dim_y << 16));   // dim0[31:16] | dim1[15:0]
  g1[3] = (int)((dim_y >> 16) | (tile_x << 16));  // dim1[31:16] | tile_dim0
  g1[4] = (int)tile_y;                            // tile_dim1; tile_dim2=0
  g1[5] = (int)stride;                            // tensor_dim0_stride[31:0]
  g1[6] = 0;
  g1[7] = 0;
  i32x4 z4 = {0, 0, 0, 0};
  i32x8 z8 = {0, 0, 0, 0, 0, 0, 0, 0};
  // 6-arg form (clang-23 / therock HIP headers lane)
  __builtin_amdgcn_tensor_load_to_lds(g0, g1, z4, z4, z8, 0);
}
#endif

// ---------------------------------------------------------------------------
// fp32 -> bf16 weight conversion
// ---------------------------------------------------------------------------
__global__ __launch_bounds__(256) void cvt_kernel(const float* __restrict__ src,
                                                  __bf16* __restrict__ dst, int n) {
  int i = blockIdx.x * 256 + threadIdx.x;
  if (i < n) dst[i] = (__bf16)src[i];
}

// ---------------------------------------------------------------------------
// Projection GEMM: out = X[M,K] @ W[N,K]^T + b, bf16 head-split epilogue.
// vmode==0 -> [B,H,L,Dk]; vmode==1 -> V transposed [B,H,Dk,L].
// ---------------------------------------------------------------------------
__global__ __launch_bounds__(256) void proj_kernel(
    const float* __restrict__ X, const __bf16* __restrict__ W,
    const float* __restrict__ bias, __bf16* __restrict__ out, int vmode) {
  const int lane = threadIdx.x & 31, wave = threadIdx.x >> 5;
  const int grp = lane >> 4, lcol = lane & 15;
  const int rm = blockIdx.y * 128 + wave * 16;
  const int cn = blockIdx.x * 64;

  v8f acc[4] = {};
  const float* arow = X + (size_t)(rm + lcol) * DMODEL;
  const __bf16* wrow = W + (size_t)(cn + lcol) * DMODEL;

  for (int ks = 0; ks < DMODEL; ks += 32) {
    __builtin_prefetch(arow + ks + 64, 0, 1);
    const int kc = ks + grp * 8;
    v16bf af = load_f32_frag(arow + kc, arow + kc + 16);
#pragma unroll
    for (int nt = 0; nt < 4; ++nt) {
      const __bf16* wb = wrow + (size_t)nt * 16 * DMODEL + kc;
      v16bf bf = load_bf16_frag(wb, wb + 16);
      acc[nt] = wmma_bf16(af, bf, acc[nt]);
    }
  }
#pragma unroll
  for (int nt = 0; nt < 4; ++nt) {
#pragma unroll
    for (int j = 0; j < 8; ++j) {
      const int gm = rm + grp * 8 + j;
      const int gn = cn + nt * 16 + lcol;
      const float val = acc[nt][j] + bias[gn];
      const int bi = gm >> 11, lq = gm & (LL - 1);
      const int h = gn >> 6, d = gn & 63;
      size_t o;
      if (vmode) o = (((size_t)(bi * NHEADS + h)) * DK + d) * LL + lq;
      else       o = (((size_t)(bi * NHEADS + h)) * LL + lq) * DK + d;
      out[o] = (__bf16)val;
    }
  }
}

// ---------------------------------------------------------------------------
// Fused flash attention. Block = 128 q rows of one (b,h); wave owns 16 rows.
// Per 32-key chunk: K-tile (4KB, contiguous) staged via async-load-to-LDS,
// V-tile (64x32, strided) staged via TDM tensor_load_to_lds; double-buffered
// so staging overlaps 8 WMMAs + online softmax. Prob tile transposed
// C-layout -> A-layout through per-wave LDS.
// ---------------------------------------------------------------------------
__global__ __launch_bounds__(256) void attn_kernel(
    const __bf16* __restrict__ q, const __bf16* __restrict__ k,
    const __bf16* __restrict__ vT, const float* __restrict__ bias,
    const int* __restrict__ mask, __bf16* __restrict__ ctx) {
  __shared__ __align__(16) __bf16 ldsK[2][32][DK];   // 8 KB: [key][dim]
  __shared__ __align__(16) __bf16 ldsV[2][DK][32];   // 8 KB: [dim][key]
  __shared__ __align__(16) __bf16 lds_p[8][16][32];  // 8 KB: per-wave P tiles
  const int tid = threadIdx.x;
  const int lane = tid & 31, wave = tid >> 5;
  const int grp = lane >> 4, lcol = lane & 15;
  const int h = blockIdx.y, b = blockIdx.z;
  const int bh = b * NHEADS + h;
  const int qbase = blockIdx.x * 128 + wave * 16;

  const __bf16* kbase = k + (size_t)bh * LL * DK;
  const __bf16* vbase = vT + (size_t)bh * DK * LL;

  const __bf16* qrow = q + ((size_t)bh * LL + qbase + lcol) * DK + grp * 8;
  v16bf qf0 = load_bf16_frag(qrow, qrow + 16);       // d = 0..31
  v16bf qf1 = load_bf16_frag(qrow + 32, qrow + 48);  // d = 32..63

  v8f acc[4] = {};
  float mrow[8], lsum[8];
#pragma unroll
  for (int j = 0; j < 8; ++j) { mrow[j] = -1e30f; lsum[j] = 0.f; }

  auto stage = [&](int kb, int nb) {
    // K tile: 32 keys x 64 dims = one contiguous 4 KB span; 1 b128/thread.
    async_copy16(&ldsK[nb][0][0] + tid * 8, kbase + (size_t)kb * DK + tid * 8);
#if HAS_TDM
    // V tile: 64 rows x 32 elems, row stride LL -> one TDM 2-D tile load.
    if (wave == 0)
      tdm_load_tile_bf16(&ldsV[nb][0][0], vbase + kb, 32u, (unsigned)DK,
                         (unsigned)LL, (unsigned)LL, (unsigned)DK);
#else
    {
      const int row = tid >> 2, c = (tid & 3) * 8;
      async_copy16(&ldsV[nb][row][c], vbase + (size_t)row * LL + kb + c);
    }
#endif
  };

  stage(0, 0);
  wait_async_lds();
#if HAS_TDM
  if (wave == 0) __builtin_amdgcn_s_wait_tensorcnt(0);
#endif
  __syncthreads();

  const float scl = 0.125f;  // 1/sqrt(Dk)
  for (int it = 0; it < LL / 32; ++it) {
    const int kb = it * 32, cur = it & 1;
    if (it + 1 < LL / 32) stage(kb + 32, cur ^ 1);

    v8f S[2];
#pragma unroll
    for (int t = 0; t < 2; ++t) {
      const __bf16* krow = &ldsK[cur][t * 16 + lcol][0];
      v16bf kf0 = load_bf16_frag(krow + grp * 8, krow + grp * 8 + 16);
      v16bf kf1 = load_bf16_frag(krow + 32 + grp * 8, krow + 48 + grp * 8);
      v8f s = {};
      s = wmma_bf16(qf0, kf0, s);
      s = wmma_bf16(qf1, kf1, s);
      S[t] = s;
    }
    // --- bias + mask + online softmax (rows m = grp*8+j, cols kb+t*16+lcol)
#pragma unroll
    for (int j = 0; j < 8; ++j) {
      const int qm = qbase + grp * 8 + j;
      const float* bp = bias + ((size_t)h * LL + qm) * LL + kb + lcol;
      const int* mp = mask + ((size_t)b * LL + qm) * LL + kb + lcol;
      float s0 = S[0][j] * scl + fminf(fmaxf(bp[0], -1e4f), 1e4f);
      float s1 = S[1][j] * scl + fminf(fmaxf(bp[16], -1e4f), 1e4f);
      if (mp[0] == 0) s0 = -1e4f;
      if (mp[16] == 0) s1 = -1e4f;
      float mx = fmaxf(s0, s1);
#pragma unroll
      for (int off = 1; off < 16; off <<= 1)
        mx = fmaxf(mx, __shfl_xor(mx, off, 32));
      const float mnew = fmaxf(mrow[j], mx);
      const float corr = __expf(mrow[j] - mnew);
      mrow[j] = mnew;
      const float p0 = __expf(s0 - mnew), p1 = __expf(s1 - mnew);
      float ls = p0 + p1;
#pragma unroll
      for (int off = 1; off < 16; off <<= 1) ls += __shfl_xor(ls, off, 32);
      lsum[j] = lsum[j] * corr + ls;
#pragma unroll
      for (int nt = 0; nt < 4; ++nt) acc[nt][j] *= corr;
      lds_p[wave][grp * 8 + j][lcol] = (__bf16)p0;
      lds_p[wave][grp * 8 + j][16 + lcol] = (__bf16)p1;
    }
    // wave-synchronous C-layout -> A-layout transpose through LDS
    asm volatile("s_wait_dscnt 0" ::: "memory");
    v16bf pf = load_bf16_frag(&lds_p[wave][lcol][grp * 8],
                              &lds_p[wave][lcol][grp * 8 + 16]);
    // --- PV: contraction over the 32 staged keys
#pragma unroll
    for (int nt = 0; nt < 4; ++nt) {
      const __bf16* vrow = &ldsV[cur][nt * 16 + lcol][0];
      v16bf vf = load_bf16_frag(vrow + grp * 8, vrow + grp * 8 + 16);
      acc[nt] = wmma_bf16(pf, vf, acc[nt]);
    }
    // retire next-chunk staging before the buffer swap
    wait_async_lds();
#if HAS_TDM
    if (wave == 0) __builtin_amdgcn_s_wait_tensorcnt(0);
#endif
    __syncthreads();
  }
  // merged-head context, bf16 [B*L, DMODEL]
#pragma unroll
  for (int nt = 0; nt < 4; ++nt) {
#pragma unroll
    for (int j = 0; j < 8; ++j) {
      const int qm = qbase + grp * 8 + j;
      const float o = acc[nt][j] / lsum[j];
      ctx[((size_t)b * LL + qm) * DMODEL + h * DK + nt * 16 + lcol] = (__bf16)o;
    }
  }
}

// ---------------------------------------------------------------------------
// Output GEMM: out = ctx[M,K](bf16) @ Wo[N,K]^T + b_o, fp32 out.
// ---------------------------------------------------------------------------
__global__ __launch_bounds__(256) void oproj_kernel(
    const __bf16* __restrict__ X, const __bf16* __restrict__ W,
    const float* __restrict__ bias, float* __restrict__ out) {
  const int lane = threadIdx.x & 31, wave = threadIdx.x >> 5;
  const int grp = lane >> 4, lcol = lane & 15;
  const int rm = blockIdx.y * 128 + wave * 16;
  const int cn = blockIdx.x * 64;

  v8f acc[4] = {};
  const __bf16* arow = X + (size_t)(rm + lcol) * DMODEL;
  const __bf16* wrow = W + (size_t)(cn + lcol) * DMODEL;

  for (int ks = 0; ks < DMODEL; ks += 32) {
    __builtin_prefetch(arow + ks + 64, 0, 1);
    const int kc = ks + grp * 8;
    v16bf af = load_bf16_frag(arow + kc, arow + kc + 16);
#pragma unroll
    for (int nt = 0; nt < 4; ++nt) {
      const __bf16* wb = wrow + (size_t)nt * 16 * DMODEL + kc;
      v16bf bf = load_bf16_frag(wb, wb + 16);
      acc[nt] = wmma_bf16(af, bf, acc[nt]);
    }
  }
#pragma unroll
  for (int nt = 0; nt < 4; ++nt) {
#pragma unroll
    for (int j = 0; j < 8; ++j) {
      const int gm = rm + grp * 8 + j;
      const int gn = cn + nt * 16 + lcol;
      out[(size_t)gm * DMODEL + gn] = acc[nt][j] + bias[gn];
    }
  }
}

// ---------------------------------------------------------------------------
extern "C" void kernel_launch(void* const* d_in, const int* in_sizes, int n_in,
                              void* d_out, int out_size, void* d_ws,
                              size_t ws_size, hipStream_t stream) {
  const float* Q    = (const float*)d_in[0];
  const float* K    = (const float*)d_in[1];
  const float* V    = (const float*)d_in[2];
  const int*   mask = (const int*)d_in[3];
  const float* bias = (const float*)d_in[4];
  const float* W_q  = (const float*)d_in[5];
  const float* b_q  = (const float*)d_in[6];
  const float* W_k  = (const float*)d_in[7];
  const float* b_k  = (const float*)d_in[8];
  const float* W_v  = (const float*)d_in[9];
  const float* b_v  = (const float*)d_in[10];
  const float* W_o  = (const float*)d_in[11];
  const float* b_o  = (const float*)d_in[12];
  float* out = (float*)d_out;

  char* ws = (char*)d_ws;
  __bf16* wqb = (__bf16*)(ws + (0ull << 20));   //  2 MB each
  __bf16* wkb = (__bf16*)(ws + (2ull << 20));
  __bf16* wvb = (__bf16*)(ws + (4ull << 20));
  __bf16* wob = (__bf16*)(ws + (6ull << 20));
  __bf16* qbuf = (__bf16*)(ws + (8ull << 20));  // 16 MB [B,H,L,Dk]
  __bf16* kbuf = (__bf16*)(ws + (24ull << 20)); // 16 MB [B,H,L,Dk]
  __bf16* vtb  = (__bf16*)(ws + (40ull << 20)); // 16 MB [B,H,Dk,L]
  __bf16* ctx  = (__bf16*)(ws + (56ull << 20)); // 16 MB [B*L, DMODEL]

  const int nw = DMODEL * DMODEL;
  dim3 cg((nw + 255) / 256);
  cvt_kernel<<<cg, 256, 0, stream>>>(W_q, wqb, nw);
  cvt_kernel<<<cg, 256, 0, stream>>>(W_k, wkb, nw);
  cvt_kernel<<<cg, 256, 0, stream>>>(W_v, wvb, nw);
  cvt_kernel<<<cg, 256, 0, stream>>>(W_o, wob, nw);

  dim3 pg(DMODEL / 64, (BB * LL) / 128);
  proj_kernel<<<pg, 256, 0, stream>>>(Q, wqb, b_q, qbuf, 0);
  proj_kernel<<<pg, 256, 0, stream>>>(K, wkb, b_k, kbuf, 0);
  proj_kernel<<<pg, 256, 0, stream>>>(V, wvb, b_v, vtb, 1);

  dim3 ag(LL / 128, NHEADS, BB);
  attn_kernel<<<ag, 256, 0, stream>>>(qbuf, kbuf, vtb, bias, mask, ctx);

  oproj_kernel<<<pg, 256, 0, stream>>>(ctx, wob, b_o, out);
}
